// ParticleFilter_53085795778860
// MI455X (gfx1250) — matrix operand order repaired and checked by reading, not compiled, not run on hardware
//
#include <hip/hip_runtime.h>
#include <hip/hip_bf16.h>
#include <math.h>

// ---------------- problem constants ----------------
#define BB 8
#define TT 50
#define PP 1024
#define SS 128
#define OO 32
#define HH 128
#define ROWS (BB * PP) // 8192 particle-rows
#define XK  (SS + OO)  // 160

typedef _Float16 f16;
typedef __attribute__((ext_vector_type(16))) _Float16 v16h;
typedef __attribute__((ext_vector_type(8)))  float    v8f;
typedef __attribute__((ext_vector_type(4)))  float    f32x4;

// ---------------- small device helpers ----------------
__device__ __forceinline__ unsigned hmix(unsigned x) {
    x ^= x >> 16; x *= 0x7feb352dU;
    x ^= x >> 15; x *= 0x846ca68bU;
    x ^= x >> 16; return x;
}
__device__ __forceinline__ float u01(unsigned h) {
    return ((h >> 8) + 0.5f) * (1.0f / 16777216.0f);
}
__device__ __forceinline__ float sigmoidf_(float x) {
    return 1.0f / (1.0f + __expf(-x));
}

// Async copy 32 bytes: global -> LDS, tracked by ASYNCcnt (no VGPR round-trip).
// ISA 15.18.3: the immediate offset is added to both the LDS and the global address.
__device__ __forceinline__ void async_copy32(unsigned lds_off, const void* gsrc) {
    asm volatile("global_load_async_to_lds_b128 %0, %1, off"
                 :: "v"(lds_off), "v"(gsrc) : "memory");
    asm volatile("global_load_async_to_lds_b128 %0, %1, off offset:16"
                 :: "v"(lds_off), "v"(gsrc) : "memory");
}
__device__ __forceinline__ void wait_async0() {
    asm volatile("s_wait_asynccnt 0" ::: "memory");
}

// A fragment (16x32 f16, row-major source, ld = K stride in halves).
// Contiguous K order: lanes 0-15 row=M, K=k0+0..15; lanes 16-31 K=k0+16..31.
__device__ __forceinline__ v16h load_a16(const f16* __restrict__ A, int ld,
                                         int row0, int k0, int lane) {
    const f16* p = A + (size_t)(row0 + (lane & 15)) * ld + k0 + ((lane >> 4) << 4);
    v16h a;
#pragma unroll
    for (int i = 0; i < 8; ++i) a[i] = p[i];
#pragma unroll
    for (int i = 0; i < 8; ++i) a[8 + i] = p[8 + i];
    return a;
}

// Read a pre-staged B fragment from LDS (fragment-native layout: 32B per lane).
__device__ __forceinline__ v16h lds_b(const f16* bsm, int f, int lane) {
    return *(const v16h*)(bsm + (size_t)f * 512 + lane * 16);
}

__device__ __forceinline__ v8f wmma32(v16h a, v16h b, v8f c) {
    return __builtin_amdgcn_wmma_f32_16x16x32_f16(false, a, false, b,
                                                  (short)0, c, false, false);
}

// ---------------- weight prep ----------------
// f32[K,N] -> transposed f16[N,K]
__global__ void k_convertT(const float* __restrict__ src, f16* __restrict__ dst,
                           int K, int N) {
    int total = K * N;
    for (int i = blockIdx.x * blockDim.x + threadIdx.x; i < total;
         i += gridDim.x * blockDim.x) {
        int n = i / K, k = i - n * K;
        dst[i] = (f16)src[k * N + n];
    }
}
// Combine W[KA,N] ++ U[KB,N] -> transposed f16[N, KA+KB]
__global__ void k_combineT(const float* __restrict__ A, int KA,
                           const float* __restrict__ B, int KB,
                           int N, f16* __restrict__ dst) {
    int LDK = KA + KB;
    int total = N * LDK;
    for (int i = blockIdx.x * blockDim.x + threadIdx.x; i < total;
         i += gridDim.x * blockDim.x) {
        int n = i / LDK, k = i - n * LDK;
        float v = (k < KA) ? A[k * N + n] : B[(k - KA) * N + n];
        dst[i] = (f16)v;
    }
}

// ---------------- state init ----------------
__global__ void k_init(float* h1s, float* c1s, float* h2s, float* c2s, float* wst) {
    int stride = gridDim.x * blockDim.x;
    int i0 = blockIdx.x * blockDim.x + threadIdx.x;
    for (int j = i0; j < ROWS * SS; j += stride) {
        h1s[j] = 0.f; c1s[j] = 0.f; h2s[j] = 0.f; c2s[j] = 0.f;
    }
    for (int j = i0; j < ROWS; j += stride) wst[j] = 1.0f / PP;
}

// ---------------- resample + input prep ----------------
// grid: (B, P/256), block 256 (8 wave32s)
__global__ void __launch_bounds__(256)
k_resample(const float* __restrict__ obs,          // [B,T,O]
           const float* __restrict__ wst,          // [B*P] logits
           const float* __restrict__ h1s, const float* __restrict__ c1s,
           const float* __restrict__ h2s, const float* __restrict__ c2s,
           f16* __restrict__ x16,                  // [ROWS,160] (noise ++ obs)
           f16* __restrict__ h1g, f16* __restrict__ h2g, // [ROWS,128] gathered f16
           float* __restrict__ c1g, float* __restrict__ c2g, // gathered f32
           int t) {
    __shared__ float wl[PP];
    __shared__ int   idxl[256];
    const int b = blockIdx.x, tile = blockIdx.y, tid = threadIdx.x;

    for (int j = tid; j < PP; j += 256) wl[j] = wst[b * PP + j];
    __syncthreads();

    { // Gumbel-max categorical over logits (== jax.random.categorical semantics)
        const int p = tile * 256 + tid;
        const unsigned base = (0x9E3779B9u * (unsigned)(t * 1315423911)) ^
                              ((unsigned)(b * PP + p) * 0x85ebca6bu) ^ 0xC0FFEE11u;
        float best = -1e30f; int bi = 0;
#pragma unroll 4
        for (int j = 0; j < PP; ++j) {
            float u = u01(hmix(base + (unsigned)j * 0x6C62272Eu));
            float g = -__logf(-__logf(u));
            float s = wl[j] + g;
            if (s > best) { best = s; bi = j; }
        }
        idxl[tid] = bi;
    }
    __syncthreads();

    const int wv = tid >> 5, lane = tid & 31;
    for (int k = 0; k < 32; ++k) { // each wave gathers 32 particles, 1 float4/lane
        const int pl  = wv * 32 + k;
        const int dst = b * PP + tile * 256 + pl;
        const int src = b * PP + idxl[pl];

        f32x4 hv1 = *(const f32x4*)(h1s + (size_t)src * SS + lane * 4);
        f32x4 hv2 = *(const f32x4*)(h2s + (size_t)src * SS + lane * 4);
        f16* d1 = h1g + (size_t)dst * SS + lane * 4;
        f16* d2 = h2g + (size_t)dst * SS + lane * 4;
        d1[0] = (f16)hv1.x; d1[1] = (f16)hv1.y; d1[2] = (f16)hv1.z; d1[3] = (f16)hv1.w;
        d2[0] = (f16)hv2.x; d2[1] = (f16)hv2.y; d2[2] = (f16)hv2.z; d2[3] = (f16)hv2.w;
        *(f32x4*)(c1g + (size_t)dst * SS + lane * 4) =
            *(const f32x4*)(c1s + (size_t)src * SS + lane * 4);
        *(f32x4*)(c2g + (size_t)dst * SS + lane * 4) =
            *(const f32x4*)(c2s + (size_t)src * SS + lane * 4);

        // x row: 128 N(0,1) noise (Box-Muller) ++ 32 obs, as f16
        f16* xr = x16 + (size_t)dst * XK;
        const unsigned nb = 0xB5297A4Du ^ (unsigned)(t * 1574811319) ^
                            ((unsigned)dst * 0x27d4eb2fu);
#pragma unroll
        for (int q = 0; q < 2; ++q) {
            unsigned hA = hmix(nb + (unsigned)(lane * 4 + q * 2) * 0x9E3779B9u);
            unsigned hB = hmix(hA ^ 0x68E31DA4u);
            float r = __fsqrt_rn(-2.0f * __logf(u01(hA)));
            float s0, c0;
            __sincosf(6.28318530718f * u01(hB), &s0, &c0);
            xr[lane * 4 + q * 2 + 0] = (f16)(r * c0);
            xr[lane * 4 + q * 2 + 1] = (f16)(r * s0);
        }
        xr[SS + lane] = (f16)obs[((size_t)b * TT + t) * OO + lane];
    }
}

// ---------------- fused LSTM layer: async double-buffered WMMA + gates ---------
// grid 64 blocks x 256 thr; wave = one 16-row M tile; block covers 128 rows.
// 16 phases = 8 n-tiles x 2 gate-pairs; phase p computes gates {2*(p&1), +1}
// of n-tile p>>1 while async-staging the next phase's fragments into LDS.
template <int KXT, int LDK>
__global__ void __launch_bounds__(256)
k_lstm(const f16* __restrict__ Xa, int ldx,        // first operand [ROWS,ldx]
       const f16* __restrict__ Ha,                 // recurrent h   [ROWS,128]
       const f16* __restrict__ CWt,                // [512, LDK] combined transposed
       const float* __restrict__ bias,             // [512]
       const float* __restrict__ Cg,               // gathered c [ROWS,128]
       float* __restrict__ Hs, float* __restrict__ Cs,
       f16* __restrict__ Hn,                       // new h as f16
       float* __restrict__ outp, int t) {          // optional particle output
    constexpr int KT   = KXT + 4;                  // total k-tiles (9 or 8)
    constexpr int NSEG = 2 * KT * 32;              // 32B segments per phase
    __shared__ f16 bsm[2][2 * KT * 512];           // 2 x (2 gates * KT frags)

    const int tid  = threadIdx.x;
    const int lane = tid & 31;
    const int wv   = tid >> 5;
    const int row0 = blockIdx.x * 128 + wv * 16;

    // Issue async staging for phase p into buffer `buf` (no wait here).
    auto stage = [&](int buf, int p) {
        const int n = p >> 1, gp = p & 1;
        for (int s = tid; s < NSEG; s += 256) {
            const int f = s >> 5, l = s & 31;
            const int g = f / KT, kk = f - g * KT;
            const int col = (gp * 2 + g) * 128 + n * 16 + (l & 15);
            const int kh  = kk * 32 + ((l >> 4) << 4);
            const f16* src = CWt + (size_t)col * LDK + kh;
            unsigned lo = (unsigned)(size_t)&bsm[buf][(size_t)f * 512 + l * 16];
            async_copy32(lo, (const void*)src);
        }
    };

    // Per-wave A fragments: KXT from Xa, 4 from Ha (K order matches CWt rows).
    v16h a[KT];
#pragma unroll
    for (int kk = 0; kk < KXT; ++kk) a[kk] = load_a16(Xa, ldx, row0, kk * 32, lane);
#pragma unroll
    for (int kk = 0; kk < 4; ++kk)
        a[KXT + kk] = load_a16(Ha, SS, row0, kk * 32, lane);

    const int colL = lane & 15;
    const int rofs = (lane >> 4) << 3;

    stage(0, 0);
    v8f acc[4];
    for (int ph = 0; ph < 16; ++ph) {
        const int buf = ph & 1, gp = ph & 1, n = ph >> 1;
        wait_async0();      // our async writes into bsm[buf] done
        __syncthreads();    // everyone's done; prior reads of bsm[buf^1] done
        if (ph < 15) stage(buf ^ 1, ph + 1);

        if (gp == 0) {      // new n-tile: zero all 4 gate accumulators
#pragma unroll
            for (int g = 0; g < 4; ++g)
#pragma unroll
                for (int r = 0; r < 8; ++r) acc[g][r] = 0.f;
        }
#pragma unroll
        for (int kk = 0; kk < KT; ++kk) {
#pragma unroll
            for (int g = 0; g < 2; ++g)
                acc[gp * 2 + g] =
                    wmma32(a[kk], lds_b(bsm[buf], g * KT + kk, lane), acc[gp * 2 + g]);
        }

        if (gp == 1) {      // all 4 gates of n-tile done: apply LSTM cell
            const int c0  = n * 16 + colL;
            const float bi_ = bias[c0],       bf_ = bias[128 + c0];
            const float bg_ = bias[256 + c0], bo_ = bias[384 + c0];
            const int col = n * 16 + colL;
#pragma unroll
            for (int r = 0; r < 8; ++r) {
                const int row = row0 + r + rofs;
                const float cold = Cg[(size_t)row * SS + col];
                const float cn = sigmoidf_(acc[1][r] + bf_) * cold +
                                 sigmoidf_(acc[0][r] + bi_) * tanhf(acc[2][r] + bg_);
                const float hn = sigmoidf_(acc[3][r] + bo_) * tanhf(cn);
                Cs[(size_t)row * SS + col] = cn;
                Hs[(size_t)row * SS + col] = hn;
                Hn[(size_t)row * SS + col] = (f16)hn;
                if (outp) {
                    const int bb_ = row >> 10, pp_ = row & (PP - 1);
                    outp[((((size_t)bb_ * TT + t) * PP + pp_)) * (SS + 1) + col] = hn;
                }
            }
        }
    }
}

// ---------------- measurement MLP ----------------
__global__ void __launch_bounds__(256)
k_measure(const float* __restrict__ obs,   // [B,T,O]
          const f16* __restrict__ H2,      // h2new f16 [ROWS,128]
          const f16* __restrict__ WmT,     // [128,160] transposed Wm1
          const float* __restrict__ bm1,   // [128]
          const float* __restrict__ Wm2,   // [128]
          const float* __restrict__ bm2,   // [1]
          float* __restrict__ wst,         // [ROWS] weights out
          float* __restrict__ outp, int t) {
    __shared__ f16 msm[40 * 512];          // all 8 n-tiles * 5 k-tiles = 40KB
    const int tid  = threadIdx.x;
    const int lane = tid & 31;
    const int wv   = tid >> 5;
    const int row0 = blockIdx.x * 128 + wv * 16;
    const int b    = row0 >> 10;

    // Async-stage the entire Wm1 fragment set once per block.
    for (int s = tid; s < 40 * 32; s += 256) {
        const int f = s >> 5, l = s & 31;
        const int n = f / 5, kk = f - n * 5;
        const int col = n * 16 + (l & 15);
        const int kh  = kk * 32 + ((l >> 4) << 4);
        const f16* src = WmT + (size_t)col * XK + kh;
        unsigned lo = (unsigned)(size_t)&msm[(size_t)f * 512 + l * 16];
        async_copy32(lo, (const void*)src);
    }

    // k-tile 0 of m_in = obs (broadcast across rows of this batch tile)
    v16h aobs;
    {
        const float* op = obs + ((size_t)b * TT + t) * OO;
        const int kb = (lane >> 4) << 4;
#pragma unroll
        for (int i = 0; i < 8; ++i) aobs[i] = (f16)op[kb + i];
#pragma unroll
        for (int i = 0; i < 8; ++i) aobs[8 + i] = (f16)op[kb + 8 + i];
    }
    v16h ah[4];
#pragma unroll
    for (int kk = 0; kk < 4; ++kk) ah[kk] = load_a16(H2, SS, row0, kk * 32, lane);

    wait_async0();
    __syncthreads();

    const int colL = lane & 15;
    float part[8];
#pragma unroll
    for (int r = 0; r < 8; ++r) part[r] = 0.f;

    for (int n = 0; n < 8; ++n) {
        const int c0 = n * 16 + colL;
        v8f acc;
#pragma unroll
        for (int r = 0; r < 8; ++r) acc[r] = 0.f;
        acc = wmma32(aobs, lds_b(msm, n * 5, lane), acc);
#pragma unroll
        for (int kk = 0; kk < 4; ++kk)
            acc = wmma32(ah[kk], lds_b(msm, n * 5 + 1 + kk, lane), acc);
        const float bv = bm1[c0];
        const float w2 = Wm2[c0];
#pragma unroll
        for (int r = 0; r < 8; ++r) {
            const float h = acc[r] + bv;
            part[r] += (h > 0.f ? h : 0.f) * w2; // ReLU then dot with Wm2
        }
    }
    // hid @ Wm2: reduce across the 16 lanes of each half-wave (rows split 0-7/8-15)
#pragma unroll
    for (int r = 0; r < 8; ++r) {
#pragma unroll
        for (int m = 8; m >= 1; m >>= 1)
            part[r] += __shfl_xor(part[r], m, 16);
    }
    if ((lane & 15) == 0) {
        const int rofs = (lane >> 4) << 3;
        const float bb2 = bm2[0];
#pragma unroll
        for (int r = 0; r < 8; ++r) {
            const int row = row0 + r + rofs;
            const float wv_ = part[r] + bb2;
            wst[row] = wv_;
            const int bb_ = row >> 10, pp_ = row & (PP - 1);
            outp[((((size_t)bb_ * TT + t) * PP + pp_)) * (SS + 1) + SS] = wv_;
        }
    }
}

// ---------------- host launcher ----------------
extern "C" void kernel_launch(void* const* d_in, const int* in_sizes, int n_in,
                              void* d_out, int out_size, void* d_ws, size_t ws_size,
                              hipStream_t stream) {
    (void)in_sizes; (void)n_in; (void)out_size; (void)ws_size;
    const float* obs = (const float*)d_in[0];
    const float* W1  = (const float*)d_in[1];
    const float* U1  = (const float*)d_in[2];
    const float* b1  = (const float*)d_in[3];
    const float* W2  = (const float*)d_in[4];
    const float* U2  = (const float*)d_in[5];
    const float* b2  = (const float*)d_in[6];
    const float* Wm1 = (const float*)d_in[7];
    const float* bm1 = (const float*)d_in[8];
    const float* Wm2 = (const float*)d_in[9];
    const float* bm2 = (const float*)d_in[10];
    float* out = (float*)d_out;

    char* base = (char*)d_ws;
    size_t off = 0;
    auto alloc = [&](size_t bytes) -> void* {
        void* p = base + off;
        off = (off + bytes + 255) & ~(size_t)255;
        return p;
    };
    f16* CW1t = (f16*)alloc((size_t)512 * (XK + SS) * 2); // [512,288]
    f16* CW2t = (f16*)alloc((size_t)512 * (SS + SS) * 2); // [512,256]
    f16* Wm1t = (f16*)alloc((size_t)HH * XK * 2);         // [128,160]
    f16* x16  = (f16*)alloc((size_t)ROWS * XK * 2);
    f16* h1g  = (f16*)alloc((size_t)ROWS * SS * 2);
    f16* h2g  = (f16*)alloc((size_t)ROWS * SS * 2);
    f16* h1n  = (f16*)alloc((size_t)ROWS * SS * 2);
    f16* h2n  = (f16*)alloc((size_t)ROWS * SS * 2);
    float* c1g = (float*)alloc((size_t)ROWS * SS * 4);
    float* c2g = (float*)alloc((size_t)ROWS * SS * 4);
    float* h1s = (float*)alloc((size_t)ROWS * SS * 4);
    float* c1s = (float*)alloc((size_t)ROWS * SS * 4);
    float* h2s = (float*)alloc((size_t)ROWS * SS * 4);
    float* c2s = (float*)alloc((size_t)ROWS * SS * 4);
    float* wst = (float*)alloc((size_t)ROWS * 4);

    // One-time per launch: combined/transposed f16 weights + zeroed states.
    k_combineT<<<256, 256, 0, stream>>>(W1, XK, U1, SS, 512, CW1t);
    k_combineT<<<256, 256, 0, stream>>>(W2, SS, U2, SS, 512, CW2t);
    k_convertT<<<64,  256, 0, stream>>>(Wm1, Wm1t, XK, HH);
    k_init<<<512, 256, 0, stream>>>(h1s, c1s, h2s, c2s, wst);

    for (int t = 0; t < TT; ++t) {
        k_resample<<<dim3(BB, PP / 256), 256, 0, stream>>>(
            obs, wst, h1s, c1s, h2s, c2s, x16, h1g, h2g, c1g, c2g, t);
        k_lstm<5, XK + SS><<<ROWS / 128, 256, 0, stream>>>(
            x16, XK, h1g, CW1t, b1, c1g, h1s, c1s, h1n, nullptr, t);
        k_lstm<4, SS + SS><<<ROWS / 128, 256, 0, stream>>>(
            h1n, SS, h2g, CW2t, b2, c2g, h2s, c2s, h2n, out, t);
        k_measure<<<ROWS / 128, 256, 0, stream>>>(
            obs, h2n, Wm1t, bm1, Wm2, bm2, wst, out, t);
    }
}
